// HaciCognitiveNet_35278861369592
// MI455X (gfx1250) — matrix operand order repaired and checked by reading, not compiled, not run on hardware
//
#include <hip/hip_runtime.h>

typedef __attribute__((ext_vector_type(16))) _Float16 v16h;
typedef __attribute__((ext_vector_type(8)))  _Float16 v8h;
typedef __attribute__((ext_vector_type(8)))  float    v8f;

#define KSTEP 64
#define AS_S  72    // A tile LDS row stride (halves): 128 rows x 64 k
#define BS_S  88    // B tile LDS col stride (halves): 64 cols x 64 k (col-major)

// ---------------------------------------------------------------------------
// WMMA GEMM:  C[z] = act( A[z] (MxK) * B[z] + bias )
//   A : f16 row-major (lda), batch offset bb*sAb + hh*sAh
//   B : f16 stored N x K (contiguous along K, leading dim ldb)
//   C : f32 (Cf) and/or f16 (Ch); transC=1 -> store at [col*ldc + row]
//   act: 0=none 1=gelu 2=sigmoid 3=retention decay mask (scale, dlog2)
// Block: 256 threads = 8 waves (4x2), tile 128x64, 32x32 per wave (2x2 WMMA),
// K-step 64 (K % 64 == 0 required).
// ---------------------------------------------------------------------------
__launch_bounds__(256)
__global__ void gemm_wmma_kernel(
    const _Float16* __restrict__ A, long long sAb, long long sAh, int lda,
    const _Float16* __restrict__ Bm, long long sBb, long long sBh, int ldb,
    const float* __restrict__ bias,
    float* __restrict__ Cf, _Float16* __restrict__ Ch, int transC,
    long long sCb, long long sCh, int ldc,
    int M, int K, int N, int H,
    int act, float scale, float dlog2)
{
    __shared__ _Float16 As[128 * AS_S];
    __shared__ _Float16 Bs[64 * BS_S];

    const int tid = threadIdx.x;
    const int z   = blockIdx.z;
    const int bb  = z / H, hh = z % H;
    const _Float16* Ap = A  + (size_t)bb * sAb + (size_t)hh * sAh;
    const _Float16* Bp = Bm + (size_t)bb * sBb + (size_t)hh * sBh;
    const size_t offC  = (size_t)bb * sCb + (size_t)hh * sCh;

    const int rowBase = blockIdx.y * 128;
    const int colBase = blockIdx.x * 64;
    const int wave = tid >> 5, lane = tid & 31;
    const int wm_ = wave & 3, wn_ = wave >> 2;      // 4 waves on M, 2 on N
    const int halfsel = lane >> 4, l16 = lane & 15;

    // A staging: 4 chunks of 8 halves per thread (rows tid>>3 + 32*j)
    const int arr = tid >> 3;                        // 0..31
    const int acc_ = (tid & 7) * 8;                  // 0..56
    size_t aoff[4]; bool aok[4];
    #pragma unroll
    for (int j = 0; j < 4; ++j) {
        const int rg = rowBase + arr + 32 * j;
        aok[j]  = rg < M;
        aoff[j] = (size_t)(aok[j] ? rg : 0) * lda;
    }
    // B staging: 2 chunks of 8 halves per thread (cols tid>>3 + 32*j)
    size_t boff[2]; bool bok[2];
    #pragma unroll
    for (int j = 0; j < 2; ++j) {
        const int cg = colBase + arr + 32 * j;
        bok[j]  = cg < N;
        boff[j] = (size_t)(bok[j] ? cg : 0) * ldb;
    }

    const v8h zero8 = {};
    v8f acc[2][2] = {};

    for (int k0 = 0; k0 < K; k0 += KSTEP) {
        // ---- stage tiles (vector loads; OOB rows/cols zeroed via select) ----
        v8h av[4], bv[2];
        #pragma unroll
        for (int j = 0; j < 4; ++j) {
            av[j] = *(const v8h*)(Ap + aoff[j] + k0 + acc_);
            if (!aok[j]) av[j] = zero8;
        }
        #pragma unroll
        for (int j = 0; j < 2; ++j) {
            bv[j] = *(const v8h*)(Bp + boff[j] + k0 + acc_);
            if (!bok[j]) bv[j] = zero8;
        }
        if (k0 + KSTEP < K) {
            __builtin_prefetch(Ap + aoff[0] + k0 + KSTEP + acc_, 0, 3);
            __builtin_prefetch(Bp + boff[0] + k0 + KSTEP + acc_, 0, 3);
        }
        #pragma unroll
        for (int j = 0; j < 4; ++j)
            *(v8h*)&As[(arr + 32 * j) * AS_S + acc_] = av[j];
        #pragma unroll
        for (int j = 0; j < 2; ++j)
            *(v8h*)&Bs[(arr + 32 * j) * BS_S + acc_] = bv[j];
        __syncthreads();

        // ---- 8 WMMAs per stage: 2 k-substeps x (2x2 register tiles) ----
        #pragma unroll
        for (int s = 0; s < 2; ++s) {
            const int kb = s * 32;
            union { v16h v; v8h h[2]; } af[2], bf[2];
            #pragma unroll
            for (int mi = 0; mi < 2; ++mi) {
                const int arow = wm_ * 32 + mi * 16 + l16;
                af[mi].h[0] = *(const v8h*)&As[arow * AS_S + kb + halfsel * 8];
                af[mi].h[1] = *(const v8h*)&As[arow * AS_S + kb + halfsel * 8 + 16];
            }
            #pragma unroll
            for (int ni = 0; ni < 2; ++ni) {
                const int bcol = wn_ * 32 + ni * 16 + l16;
                bf[ni].h[0] = *(const v8h*)&Bs[bcol * BS_S + kb + halfsel * 8];
                bf[ni].h[1] = *(const v8h*)&Bs[bcol * BS_S + kb + halfsel * 8 + 16];
            }
            #pragma unroll
            for (int mi = 0; mi < 2; ++mi)
                #pragma unroll
                for (int ni = 0; ni < 2; ++ni)
                    acc[mi][ni] = __builtin_amdgcn_wmma_f32_16x16x32_f16(
                        false, af[mi].v, false, bf[ni].v, (short)0,
                        acc[mi][ni], false, false);
        }
        __syncthreads();
    }

    // ---- epilogue: C layout M = i + 8*halfsel (VGPR i), N = lane%16 ----
    #pragma unroll
    for (int mi = 0; mi < 2; ++mi) {
        #pragma unroll
        for (int ni = 0; ni < 2; ++ni) {
            const int cg = colBase + wn_ * 32 + ni * 16 + l16;
            #pragma unroll
            for (int i = 0; i < 8; ++i) {
                const int rg = rowBase + wm_ * 32 + mi * 16 + halfsel * 8 + i;
                if (rg < M && cg < N) {
                    float v = acc[mi][ni][i];
                    if (bias) v += bias[cg];
                    if (act == 1) {
                        v = 0.5f * v * (1.0f + erff(v * 0.70710678118f));
                    } else if (act == 2) {
                        v = 1.0f / (1.0f + __expf(-v));
                    } else if (act == 3) {
                        const int d = rg - cg;      // s - t
                        v = (d >= 0) ? v * scale * exp2f(dlog2 * (float)d) : 0.0f;
                    }
                    const size_t o = transC ? offC + (size_t)cg * ldc + rg
                                            : offC + (size_t)rg * ldc + cg;
                    if (Cf) Cf[o] = v;
                    if (Ch) Ch[o] = (_Float16)v;
                }
            }
        }
    }
}

// ---------------------------------------------------------------------------
// LayerNorm (one block per row) with optional f32 and f16 outputs
// ---------------------------------------------------------------------------
__launch_bounds__(256)
__global__ void ln_kernel(const float* __restrict__ x,
                          const float* __restrict__ g,
                          const float* __restrict__ b,
                          float* __restrict__ outf, _Float16* __restrict__ outh, int d)
{
    __shared__ float red[256];
    const int row = blockIdx.x;
    const float* xr = x + (size_t)row * d;
    float s = 0.f, s2 = 0.f;
    for (int i = threadIdx.x; i < d; i += 256) {
        float v = xr[i]; s += v; s2 += v * v;
    }
    red[threadIdx.x] = s; __syncthreads();
    for (int st = 128; st > 0; st >>= 1) {
        if (threadIdx.x < st) red[threadIdx.x] += red[threadIdx.x + st];
        __syncthreads();
    }
    const float mean = red[0] / (float)d;
    __syncthreads();
    red[threadIdx.x] = s2; __syncthreads();
    for (int st = 128; st > 0; st >>= 1) {
        if (threadIdx.x < st) red[threadIdx.x] += red[threadIdx.x + st];
        __syncthreads();
    }
    const float var  = red[0] / (float)d - mean * mean;
    const float rstd = rsqrtf(var + 1e-5f);
    for (int i = threadIdx.x; i < d; i += 256) {
        float v = (xr[i] - mean) * rstd * g[i] + b[i];
        if (outf) outf[(size_t)row * d + i] = v;
        if (outh) outh[(size_t)row * d + i] = (_Float16)v;
    }
}

// ---------------------------------------------------------------------------
// Elementwise helpers
// ---------------------------------------------------------------------------
__global__ void f2h_kernel(const float* __restrict__ x, _Float16* __restrict__ y, size_t n)
{
    size_t i = (size_t)blockIdx.x * 256 + threadIdx.x;
    if (i < n) y[i] = (_Float16)x[i];
}
// transpose-convert: src f32 (K x N) -> dst f16 (N x K)
__global__ void f2h_t_kernel(const float* __restrict__ src, _Float16* __restrict__ dst,
                             int Kd, int Nd)
{
    size_t i = (size_t)blockIdx.x * 256 + threadIdx.x;
    if (i < (size_t)Kd * Nd) {
        int k = (int)(i / Nd), n = (int)(i % Nd);
        dst[(size_t)n * Kd + k] = (_Float16)src[i];
    }
}
__global__ void add_kernel(float* __restrict__ x, const float* __restrict__ y, size_t n)
{
    size_t i = (size_t)blockIdx.x * 256 + threadIdx.x;
    if (i < n) x[i] += y[i];
}
// o16 = (f16)(a * g)
__global__ void gatemul_kernel(_Float16* __restrict__ o16, const float* __restrict__ a,
                               const float* __restrict__ g, size_t n)
{
    size_t i = (size_t)blockIdx.x * 256 + threadIdx.x;
    if (i < n) o16[i] = (_Float16)(a[i] * g[i]);
}
__global__ void pool_kernel(const float* __restrict__ ws, float* __restrict__ pooled)
{
    const int b = blockIdx.x;
    for (int d = threadIdx.x; d < 512; d += 256) {
        float s = 0.f;
        for (int t = 0; t < 512; ++t)
            s += ws[((size_t)(b * 512 + t)) * 512 + d];
        pooled[b * 512 + d] = s * (1.0f / 512.0f);
    }
}
__global__ void gate_kernel(const float* __restrict__ pooled, float* __restrict__ gate)
{
    __shared__ float red[256];
    const int b = blockIdx.x;
    float s = 0.f;
    for (int d = threadIdx.x; d < 512; d += 256) {
        float v = pooled[b * 512 + d]; s += v * v;
    }
    red[threadIdx.x] = s; __syncthreads();
    for (int st = 128; st > 0; st >>= 1) {
        if (threadIdx.x < st) red[threadIdx.x] += red[threadIdx.x + st];
        __syncthreads();
    }
    if (threadIdx.x == 0) {
        float nrm = sqrtf(red[0]) * (1.0f / 512.0f);
        gate[b] = 1.0f / (1.0f + __expf(-nrm));
    }
}
__global__ void persflat_kernel(const float* __restrict__ vec, const float* __restrict__ gate,
                                float* __restrict__ flat)
{
    int i = blockIdx.x * 256 + threadIdx.x;
    if (i < 8 * 1024) {
        int b = i >> 10;
        flat[i] = vec[i & 1023] * (0.5f + 0.5f * gate[b]);
    }
}
__global__ void concat_kernel(const float* __restrict__ ws, const float* __restrict__ ps,
                              float* __restrict__ x)
{
    size_t i = (size_t)blockIdx.x * 256 + threadIdx.x;
    const size_t n = (size_t)4096 * 1536;
    if (i < n) {
        int t = (int)(i / 1536), c = (int)(i % 1536), b = t >> 9;
        x[i] = (c < 512) ? ws[(size_t)t * 512 + c] : ps[b * 1024 + (c - 512)];
    }
}
__global__ void last_kernel(const float* __restrict__ y, float* __restrict__ last)
{
    int i = blockIdx.x * 256 + threadIdx.x;
    if (i < 8 * 1536) {
        int b = i / 1536, c = i % 1536;
        last[i] = y[((size_t)(b * 512 + 511)) * 1536 + c];
    }
}

// ---------------------------------------------------------------------------
// Host orchestration
// ---------------------------------------------------------------------------
static inline int nblk(size_t n) { return (int)((n + 255) / 256); }

static void launch_gemm(hipStream_t st,
    const _Float16* A, long long sAb, long long sAh, int lda,
    const _Float16* Bm, long long sBb, long long sBh, int ldb,
    const float* bias, float* Cf, _Float16* Ch, int transC,
    long long sCb, long long sCh, int ldc,
    int M, int K, int N, int H, int Z, int act, float scale, float dlog2)
{
    dim3 grid((N + 63) / 64, (M + 127) / 128, Z);
    gemm_wmma_kernel<<<grid, 256, 0, st>>>(A, sAb, sAh, lda, Bm, sBb, sBh, ldb,
                                           bias, Cf, Ch, transC, sCb, sCh, ldc,
                                           M, K, N, H, act, scale, dlog2);
}
static void gemm_plain(hipStream_t st, const _Float16* A, int lda,
                       const _Float16* Bm, int ldb,
                       const float* bias, float* Cf, _Float16* Ch, int transC, int ldc,
                       int M, int K, int N, int act)
{
    launch_gemm(st, A, 0, 0, lda, Bm, 0, 0, ldb, bias, Cf, Ch, transC, 0, 0, ldc,
                M, K, N, 1, 1, act, 1.0f, 0.0f);
}

extern "C" void kernel_launch(void* const* d_in, const int* in_sizes, int n_in,
                              void* d_out, int out_size, void* d_ws, size_t ws_size,
                              hipStream_t stream)
{
    (void)in_sizes; (void)n_in; (void)out_size; (void)ws_size;
    const int B = 8, S = 512, T = B * S, IN = 384, WD = 512, PD = 1024, TD = WD + PD, H = 8;
    const float DLOG2 = -0.0144995696f;   // log2(0.99)

    auto L = [&](int i) -> const float* { return (const float*)d_in[i]; };
    const float* mem      = L(0);
    const float* conf_b   = L(1);  const float* conf_w  = L(2);
    const float* fus1_b   = L(3);  const float* fus1_w  = L(4);
    const float* fus2_b   = L(5);  const float* fus2_w  = L(6);
    const float* ins_b    = L(7);  const float* ins_w   = L(8);
    const float* outln_b  = L(9);  const float* outln_g = L(10);
    const float* pln_b    = L(11); const float* pln_g   = L(12);
    const float* pproj_b  = L(13); const float* pproj_w = L(14);
    const float* pvec     = L(15);
    const float* pred_b   = L(24); const float* pred_w  = L(25);
    const float* ffn1_b   = L(74); const float* ffn1_w  = L(75);
    const float* ffn2_b   = L(76); const float* ffn2_w  = L(77);
    const float* inp_b    = L(78); const float* inp_w   = L(79);
    const float* wmoln_b  = L(80); const float* wmoln_g = L(81);
    // pre_ln[i]: b=16+2i g=17+2i ; ret[i] base=26+12i
    // wm.pre_ln[i]: b=82+2i g=83+2i ; wm.ret[i] base=86+12i
    // ret dict: +0 g.b +1 g.w +2 k.b +3 k.w +4 ln_b +5 ln_g +6 o.b +7 o.w +8 q.b +9 q.w +10 v.b +11 v.w

    char* p = (char*)d_ws;
    auto carve = [&](size_t bytes) -> char* {
        char* r = p; p += (bytes + 255) & ~(size_t)255; return r;
    };
    float*    X    = (float*)   carve((size_t)T * TD * 4);
    float*    Y    = (float*)   carve((size_t)T * TD * 4);
    float*    Zb   = (float*)   carve((size_t)T * TD * 4);
    float*    RB   = (float*)   carve((size_t)T * TD * 4);
    _Float16* A16  = (_Float16*)carve((size_t)T * 2048 * 2);
    _Float16* H16  = (_Float16*)carve((size_t)T * 2048 * 2);
    _Float16* Q16  = (_Float16*)carve((size_t)T * TD * 2);
    _Float16* K16  = (_Float16*)carve((size_t)T * TD * 2);
    _Float16* Vt16 = (_Float16*)carve((size_t)T * TD * 2);      // V stored D x T
    _Float16* S16  = (_Float16*)carve((size_t)B * H * S * S * 2);
    _Float16* W16  = (_Float16*)carve((size_t)2359296 * 2);     // weights, N x K
    float*    SMALL = (float*)  carve(65536 * 4);
    float* pooled  = SMALL;
    float* gateb   = pooled + B * WD;
    float* flat    = gateb + 64;
    float* projout = flat + B * PD;
    float* pstate  = projout + B * PD;
    float* lastv   = pstate + B * PD;

    auto cv  = [&](const float* s, _Float16* d, size_t n) {
        f2h_kernel<<<nblk(n), 256, 0, stream>>>(s, d, n);
    };
    auto cvt = [&](const float* s, _Float16* d, int Kd, int Nd) {
        f2h_t_kernel<<<nblk((size_t)Kd * Nd), 256, 0, stream>>>(s, d, Kd, Nd);
    };
    auto add = [&](float* x, const float* y, size_t n) {
        add_kernel<<<nblk(n), 256, 0, stream>>>(x, y, n);
    };
    auto lnv = [&](const float* x, const float* g, const float* b,
                   float* of, _Float16* oh, int rows, int d) {
        ln_kernel<<<rows, 256, 0, stream>>>(x, g, b, of, oh, d);
    };

    auto retention = [&](int rb, int pg_i, int pb_i, int D) {
        const int dh = D / H;
        lnv(X, L(pg_i), L(pb_i), nullptr, A16, T, D);
        cvt(L(rb + 9),  W16, D, D);
        gemm_plain(stream, A16, D, W16, D, L(rb + 8),  nullptr, Q16,  0, D, T, D, D, 0);
        cvt(L(rb + 3),  W16, D, D);
        gemm_plain(stream, A16, D, W16, D, L(rb + 2),  nullptr, K16,  0, D, T, D, D, 0);
        cvt(L(rb + 11), W16, D, D);
        gemm_plain(stream, A16, D, W16, D, L(rb + 10), nullptr, Vt16, 1, T, T, D, D, 0);
        cvt(L(rb + 1),  W16, D, D);
        gemm_plain(stream, A16, D, W16, D, L(rb + 0),  Zb, nullptr,   0, D, T, D, D, 2);
        launch_gemm(stream,
                    Q16, (long long)S * D, dh, D,
                    K16, (long long)S * D, dh, D,
                    nullptr, nullptr, S16, 0,
                    (long long)H * S * S, (long long)S * S, S,
                    S, dh, S, H, B * H, 3, 1.0f / sqrtf((float)dh), DLOG2);
        launch_gemm(stream,
                    S16, (long long)H * S * S, (long long)S * S, S,
                    Vt16, (long long)S, (long long)dh * T, T,
                    nullptr, RB, nullptr, 0,
                    (long long)S * D, dh, D,
                    S, S, dh, H, B * H, 0, 1.0f, 0.0f);
        lnv(RB, L(rb + 5), L(rb + 4), Y, nullptr, T, D);
        gatemul_kernel<<<nblk((size_t)T * D), 256, 0, stream>>>(A16, Y, Zb, (size_t)T * D);
        cvt(L(rb + 7), W16, D, D);
        gemm_plain(stream, A16, D, W16, D, L(rb + 6), RB, nullptr, 0, D, T, D, D, 0);
        add(X, RB, (size_t)T * D);
    };

    // ================= Phase A: world model (D = 512) =================
    cv(mem, A16, (size_t)T * IN);
    cvt(inp_w, W16, IN, WD);
    gemm_plain(stream, A16, IN, W16, IN, inp_b, X, nullptr, 0, WD, T, IN, WD, 0);

    for (int l = 0; l < 2; ++l) {
        retention(86 + 12 * l, 83 + 2 * l, 82 + 2 * l, WD);
        cv(X, A16, (size_t)T * WD);
        cvt(ffn1_w, W16, WD, 2048);
        gemm_plain(stream, A16, WD, W16, WD, ffn1_b, nullptr, H16, 0, 2048, T, WD, 2048, 1);
        cvt(ffn2_w, W16, 2048, WD);
        gemm_plain(stream, H16, 2048, W16, 2048, ffn2_b, RB, nullptr, 0, WD, T, 2048, WD, 0);
        add(X, RB, (size_t)T * WD);
    }
    lnv(X, wmoln_g, wmoln_b, Y, nullptr, T, WD);                 // Y = world_state

    // ================= Personality branch =================
    pool_kernel<<<B, 256, 0, stream>>>(Y, pooled);
    gate_kernel<<<B, 256, 0, stream>>>(pooled, gateb);
    persflat_kernel<<<nblk(B * PD), 256, 0, stream>>>(pvec, gateb, flat);
    cv(flat, A16, (size_t)B * PD);
    cvt(pproj_w, W16, PD, PD);
    gemm_plain(stream, A16, PD, W16, PD, pproj_b, projout, nullptr, 0, PD, B, PD, PD, 0);
    lnv(projout, pln_g, pln_b, pstate, nullptr, B, PD);
    concat_kernel<<<nblk((size_t)T * TD), 256, 0, stream>>>(Y, pstate, X);

    // ================= Phase B: fusion stack (D = 1536) =================
    for (int l = 0; l < 4; ++l)
        retention(26 + 12 * l, 17 + 2 * l, 16 + 2 * l, TD);

    // fus: x = fus2(gelu(fus1(x))) + x
    cv(X, A16, (size_t)T * TD);
    cvt(fus1_w, W16, TD, TD);
    gemm_plain(stream, A16, TD, W16, TD, fus1_b, nullptr, H16, 0, TD, T, TD, TD, 1);
    cvt(fus2_w, W16, TD, TD);
    gemm_plain(stream, H16, TD, W16, TD, fus2_b, RB, nullptr, 0, TD, T, TD, TD, 0);
    add(X, RB, (size_t)T * TD);
    lnv(X, outln_g, outln_b, Y, nullptr, T, TD);

    // ================= Heads =================
    last_kernel<<<nblk(B * TD), 256, 0, stream>>>(Y, lastv);
    cv(lastv, A16, (size_t)B * TD);
    float* out_f = (float*)d_out;
    cvt(pred_w, W16, TD, IN);
    gemm_plain(stream, A16, TD, W16, TD, pred_b, out_f, nullptr, 0, IN, B, TD, IN, 0);
    cvt(ins_w, W16, TD, WD);
    gemm_plain(stream, A16, TD, W16, TD, ins_b, out_f + B * IN, nullptr, 0, WD, B, TD, WD, 0);
    cvt(conf_w, W16, TD, 1);
    gemm_plain(stream, A16, TD, W16, TD, conf_b, out_f + B * IN + B * WD, nullptr, 0, 1,
               B, TD, 1, 2);
}
